// Batch_Edge_21079699489178
// MI455X (gfx1250) — matrix operand order repaired and checked by reading, not compiled, not run on hardware
//
#include <hip/hip_runtime.h>
#include <math.h>

typedef __attribute__((ext_vector_type(16))) _Float16 v16h;
typedef __attribute__((ext_vector_type(8)))  _Float16 v8h;
typedef __attribute__((ext_vector_type(4)))  _Float16 v4h;
typedef __attribute__((ext_vector_type(8)))  float    v8f;
typedef __attribute__((ext_vector_type(4)))  float    v4f;
typedef __attribute__((ext_vector_type(2)))  float    v2f;

#define HID   128
#define TWOH  256
#define BGRAPH 512
#define NNODES (512L * 256L)
#define ROWS_PER_WAVE 32
#define WAVES_PER_WG  2
#define LDS_ROW 264   // halves per row, padded (+8) to kill LDS bank conflicts

// Branch-free tanh: tanh(x) = sign(x) * (1 - e^{-2|x|}) / (1 + e^{-2|x|}).
// exp argument is always <= 0 -> no overflow; fully branchless (native
// v_exp_f32 + v_rcp_f32), co-executes with XDL WMMA.
__device__ inline float fast_tanh(float x) {
  float ax = __builtin_fabsf(x);
  float t  = __expf(-2.0f * ax);
  float r  = (1.0f - t) * __builtin_amdgcn_rcpf(1.0f + t);
  return __builtin_copysignf(r, x);
}

// ---------------------------------------------------------------------------
// Stage 0: node_embed = last @ W_embed + b_embed  (fp32 compute, f16 out)
// ---------------------------------------------------------------------------
__global__ __launch_bounds__(128)
void embed_kernel(const float* __restrict__ last, const float* __restrict__ We,
                  const float* __restrict__ be, _Float16* __restrict__ outE) {
  int b = blockIdx.x, n = threadIdx.x;
  const float* lrow = last + (long)b * HID;
  float acc = be[n];
#pragma unroll 4
  for (int k = 0; k < HID; ++k) acc += lrow[k] * We[k * HID + n];
  outE[(long)b * HID + n] = (_Float16)acc;
}

// ---------------------------------------------------------------------------
// B-fragment packing for v_wmma_f32_16x16x32_f16.
// ISA 16-bit B layout (32x16, 8 VGPRs = 16 halves/lane):
//   lane L holds column N = L%16; halves 0..7 -> K = (L/16)*8 + j,
//   halves 8..15 -> K = 16 + (L/16)*8 + (j-8).
// Fragment (kb, nb) stored contiguously: 32 lanes x 16 halves = 512 halves.
// ---------------------------------------------------------------------------
__device__ __host__ inline int bfrag_index(int k, int n) {
  int kb = k >> 5, nb = n >> 4;
  int lane = (n & 15) | (((k >> 3) & 1) << 4);
  int j = (k & 7) | (((k >> 4) & 1) << 3);
  return ((nb * 8 + kb) * 32 + lane) * 16 + j;
}

__global__ __launch_bounds__(256)
void pack_w_kernel(const float* __restrict__ W1, const float* __restrict__ W2,
                   _Float16* __restrict__ p1, _Float16* __restrict__ p2) {
  int k = blockIdx.x;        // 0..255 (input dim)
  int n = threadIdx.x;       // 0..255 (output dim)
  int d = bfrag_index(k, n);
  p1[d] = (_Float16)W1[k * TWOH + n];
  p2[d] = (_Float16)W2[k * TWOH + n];
}

__global__ __launch_bounds__(256)
void pack_w3_kernel(const float* __restrict__ W3, _Float16* __restrict__ p3) {
  int idx = blockIdx.x * 256 + threadIdx.x;   // 4096 = 256 k * 16 n
  int k = idx >> 4, n = idx & 15;
  int d = bfrag_index(k, n);                   // nb == 0
  p3[d] = (n < 2) ? (_Float16)W3[k * 2 + n] : (_Float16)0.0f;
}

// ---------------------------------------------------------------------------
// Fused MLP: stack=[h|embed] -> tanh(W1) -> tanh(W2) -> W3, all via WMMA.
// 2 waves per WG, 32 rows per wave; LDS tile is reused between layers
// (A fragments are fully register-resident before outputs overwrite LDS;
//  same-wave DS ops are in-order).
// ---------------------------------------------------------------------------
__global__ __launch_bounds__(WAVES_PER_WG * 32)
void mlp_kernel(const float* __restrict__ h, const _Float16* __restrict__ embedE,
                const _Float16* __restrict__ pW1, const _Float16* __restrict__ pW2,
                const _Float16* __restrict__ pW3,
                const float* __restrict__ b1, const float* __restrict__ b2,
                const float* __restrict__ b3, float* __restrict__ out) {
  __shared__ __align__(16) _Float16 lds[WAVES_PER_WG][ROWS_PER_WAVE][LDS_ROW];
  const int wave = threadIdx.x >> 5;
  const int lane = threadIdx.x & 31;
  const long rowBase = ((long)blockIdx.x * WAVES_PER_WG + wave) * ROWS_PER_WAVE;
  _Float16* T = &lds[wave][0][0];

  // ---- stage stack tile = [ h (f32->f16) | graph embed (f16) ] ----
  {
    const long g = rowBase >> 8;   // 256 nodes per graph; 32-row block never crosses
    v4h ev = *(const v4h*)(embedE + g * HID + lane * 4);
#pragma unroll 4
    for (int r = 0; r < ROWS_PER_WAVE; ++r) {
      v4f hv = *(const v4f*)(h + (rowBase + r) * HID + lane * 4);
      v4h hh;
#pragma unroll
      for (int j = 0; j < 4; ++j) hh[j] = (_Float16)hv[j];
      *(v4h*)(T + r * LDS_ROW + lane * 4) = hh;
      *(v4h*)(T + r * LDS_ROW + HID + lane * 4) = ev;
    }
  }

  // A-fragment loader per ISA 16-bit A layout (16x32):
  // lane L: M = L%16; halves 0..7 -> K = kb*32 + (L/16)*8 + j
  //                   halves 8..15 -> K = kb*32 + 16 + (L/16)*8 + j
  auto loadA = [&](int s, int kb) -> v16h {
    const _Float16* p = T + (s * 16 + (lane & 15)) * LDS_ROW + kb * 32 + (lane >> 4) * 8;
    v8h lo = *(const v8h*)p;
    v8h hi = *(const v8h*)(p + 16);
    v16h a;
#pragma unroll
    for (int j = 0; j < 8; ++j) { a[j] = lo[j]; a[j + 8] = hi[j]; }
    return a;
  };

  v16h A[2][8];

  // ---- layers 1 and 2 (tanh) ----
#pragma unroll 1
  for (int layer = 0; layer < 2; ++layer) {
#pragma unroll
    for (int s = 0; s < 2; ++s)
#pragma unroll
      for (int kb = 0; kb < 8; ++kb) A[s][kb] = loadA(s, kb);

    const _Float16* Wp = layer ? pW2 : pW1;
    const float*    bp = layer ? b2 : b1;

    // preload all biases this lane needs (one load clause per layer)
    float bn[16];
#pragma unroll
    for (int nb = 0; nb < 16; ++nb) bn[nb] = bp[nb * 16 + (lane & 15)];

#pragma unroll 1
    for (int nb = 0; nb < 16; ++nb) {
      // Preload ALL 8 B fragments for this n-tile into registers so the
      // load clause pipelines against the WMMA chain (no per-kb stalls).
      v16h Bf[8];
#pragma unroll
      for (int kb = 0; kb < 8; ++kb)
        Bf[kb] = *(const v16h*)(Wp + (((long)(nb * 8 + kb)) * 32 + lane) * 16);

      v8f c0, c1;
#pragma unroll
      for (int v = 0; v < 8; ++v) { c0[v] = bn[nb]; c1[v] = bn[nb]; }
#pragma unroll
      for (int kb = 0; kb < 8; ++kb) {
        c0 = __builtin_amdgcn_wmma_f32_16x16x32_f16(false, A[0][kb], false, Bf[kb],
                                                    (short)0, c0, false, false);
        c1 = __builtin_amdgcn_wmma_f32_16x16x32_f16(false, A[1][kb], false, Bf[kb],
                                                    (short)0, c1, false, false);
      }
      // D layout: lane L holds N=L%16, rows M = v + 8*(L/16)
      const int n  = nb * 16 + (lane & 15);
      const int rb = (lane >> 4) * 8;
#pragma unroll
      for (int v = 0; v < 8; ++v) {
        T[(rb + v) * LDS_ROW + n]      = (_Float16)fast_tanh(c0[v]);
        T[(16 + rb + v) * LDS_ROW + n] = (_Float16)fast_tanh(c1[v]);
      }
    }
  }

  // ---- layer 3: [N,256] @ W3(256x2, zero-padded to x16) ----
  {
#pragma unroll
    for (int s = 0; s < 2; ++s)
#pragma unroll
      for (int kb = 0; kb < 8; ++kb) A[s][kb] = loadA(s, kb);

    v16h Bf[8];
#pragma unroll
    for (int kb = 0; kb < 8; ++kb)
      Bf[kb] = *(const v16h*)(pW3 + ((long)kb * 32 + lane) * 16);

    const int n = lane & 15;
    const float bnv = (n < 2) ? b3[n] : 0.0f;
    v8f c0, c1;
#pragma unroll
    for (int v = 0; v < 8; ++v) { c0[v] = bnv; c1[v] = bnv; }
#pragma unroll
    for (int kb = 0; kb < 8; ++kb) {
      c0 = __builtin_amdgcn_wmma_f32_16x16x32_f16(false, A[0][kb], false, Bf[kb],
                                                  (short)0, c0, false, false);
      c1 = __builtin_amdgcn_wmma_f32_16x16x32_f16(false, A[1][kb], false, Bf[kb],
                                                  (short)0, c1, false, false);
    }

    // Transpose the 2 live output columns through (now dead) LDS so the
    // global write is one coalesced 8B/lane burst (256B contiguous per wave).
    float* Tf = (float*)T;           // reuse stack tile; 16B aligned
    if (n < 2) {                     // divergence only after all WMMAs
      const int rb = (lane >> 4) * 8;
#pragma unroll
      for (int v = 0; v < 8; ++v) {
        Tf[(rb + v) * 2 + n]      = c0[v];   // rows 0..15
        Tf[(16 + rb + v) * 2 + n] = c1[v];   // rows 16..31
      }
    }
    // same-wave DS ops are in-order: load sees the stores above
    v2f e = *(const v2f*)(Tf + lane * 2);
    *(v2f*)(out + (rowBase + lane) * 2) = e;
  }
}

// ---------------------------------------------------------------------------
extern "C" void kernel_launch(void* const* d_in, const int* in_sizes, int n_in,
                              void* d_out, int out_size, void* d_ws, size_t ws_size,
                              hipStream_t stream) {
  const float* last = (const float*)d_in[0];
  const float* h    = (const float*)d_in[1];
  const float* We   = (const float*)d_in[2];
  const float* be   = (const float*)d_in[3];
  const float* W1   = (const float*)d_in[4];
  const float* b1   = (const float*)d_in[5];
  const float* W2   = (const float*)d_in[6];
  const float* b2   = (const float*)d_in[7];
  const float* W3   = (const float*)d_in[8];
  const float* b3   = (const float*)d_in[9];
  float* out = (float*)d_out;

  // workspace layout (halves): embed | packed W1 | packed W2 | packed W3(pad)
  _Float16* wsE  = (_Float16*)d_ws;                 // 512*128
  _Float16* wsW1 = wsE + (long)BGRAPH * HID;        // 256*256
  _Float16* wsW2 = wsW1 + (long)TWOH * TWOH;        // 256*256
  _Float16* wsW3 = wsW2 + (long)TWOH * TWOH;        // 256*16

  embed_kernel<<<BGRAPH, HID, 0, stream>>>(last, We, be, wsE);
  pack_w_kernel<<<TWOH, TWOH, 0, stream>>>(W1, W2, wsW1, wsW2);
  pack_w3_kernel<<<16, 256, 0, stream>>>(W3, wsW3);

  const int blocks = (int)(NNODES / (WAVES_PER_WG * ROWS_PER_WAVE));  // 2048
  mlp_kernel<<<blocks, WAVES_PER_WG * 32, 0, stream>>>(
      h, wsE, wsW1, wsW2, wsW3, b1, b2, b3, out);
}
